// EdgeMLPDecoder_996432413121
// MI455X (gfx1250) — matrix operand order repaired and checked by reading, not compiled, not run on hardware
//
#include <hip/hip_runtime.h>
#include <hip/hip_bf16.h>
#include <math.h>

typedef __attribute__((ext_vector_type(16))) __bf16 v16bf;
typedef __attribute__((ext_vector_type(8)))  __bf16 v8bf;
typedef __attribute__((ext_vector_type(4)))  __bf16 v4bf;
typedef __attribute__((ext_vector_type(8)))  float  v8f;
typedef __attribute__((ext_vector_type(4)))  float  v4f;

#define DD    128        // feature dim D
#define K1    256        // 2*D   (GEMM1 K)
#define K1P   264        // padded row (bank-conflict avoidance)
#define HIDN  64         // hidden 1
#define HIDP  72         // padded
#define HID2  32         // hidden 2
#define NROW  268        // N
#define WAVES 8
#define ROWS  16         // edges per wave
#define EPB   (WAVES*ROWS)   // 128 edges per block
#define SHP   68         // padded f32 scratch row

union AF { v16bf v; v8bf h[2]; };

__device__ __forceinline__ float gelu_exact(float x) {
    return 0.5f * x * (1.0f + erff(x * 0.70710678118654752440f));
}

__global__ __launch_bounds__(256, 1)
void edge_mlp_wmma_kernel(const float* __restrict__ H,
                          const int*   __restrict__ idx,
                          const float* __restrict__ W1, const float* __restrict__ b1,
                          const float* __restrict__ g1, const float* __restrict__ be1,
                          const float* __restrict__ W2, const float* __restrict__ b2,
                          const float* __restrict__ g2, const float* __restrict__ be2,
                          const float* __restrict__ W3, const float* __restrict__ b3,
                          float* __restrict__ out, int E)
{
    // ---- LDS staging ----
    __shared__ __align__(16) __bf16 sW1t[HIDN][K1P];          // W1^T  64x256 bf16
    __shared__ __align__(16) __bf16 sW2t[HID2][HIDP];         // W2^T  32x64  bf16
    __shared__ __align__(16) __bf16 sE[WAVES][ROWS][K1P];     // edge features bf16
    __shared__ __align__(16) __bf16 sA[WAVES][ROWS][HIDP];    // post-act h1 bf16
    __shared__ __align__(16) float  sH[WAVES][ROWS][SHP];     // f32 scratch (h1 pre-LN / h2)
    __shared__ float sB1[HIDN], sG1[HIDN], sBe1[HIDN];
    __shared__ float sB2[HID2], sG2[HID2], sBe2[HID2], sW3[HID2];
    __shared__ float sB3;

    const int tid  = threadIdx.x;
    const int wave = tid >> 5;
    const int lane = tid & 31;
    const int bidx = blockIdx.y;

    // ---- one-time weight preload (transposed, bf16) ----
    for (int i = tid; i < HIDN * K1; i += 256) {
        const int n = i >> 8;            // output channel
        const int k = i & 255;           // K index (contiguous per thread run)
        sW1t[n][k] = (__bf16)W1[k * HIDN + n];
    }
    for (int i = tid; i < HID2 * HIDN; i += 256) {
        const int n = i >> 6;
        const int k = i & 63;
        sW2t[n][k] = (__bf16)W2[k * HID2 + n];
    }
    if (tid < HIDN) { sB1[tid] = b1[tid]; sG1[tid] = g1[tid]; sBe1[tid] = be1[tid]; }
    if (tid < HID2) { sB2[tid] = b2[tid]; sG2[tid] = g2[tid]; sBe2[tid] = be2[tid]; sW3[tid] = W3[tid]; }
    if (tid == 0)   sB3 = b3[0];
    __syncthreads();

    // ---- build 16x256 bf16 edge-feature tile (per wave) ----
    const int ebase = blockIdx.x * EPB + wave * ROWS;
    const float* __restrict__ Hb = H + (size_t)bidx * NROW * DD;
    #pragma unroll 4
    for (int r = 0; r < ROWS; ++r) {
        int e = ebase + r; if (e >= E) e = E - 1;
        const int i0 = idx[e];
        const int j0 = idx[E + e];
        const float* U = Hb + (size_t)i0 * DD;
        const float* V = Hb + (size_t)j0 * DD;
        const int c = lane * 4;                       // 32 lanes x 4 cols = 128
        const v4f u = *(const v4f*)(U + c);
        const v4f v = *(const v4f*)(V + c);
        v4bf p, s;
        #pragma unroll
        for (int q = 0; q < 4; ++q) {
            p[q] = (__bf16)(u[q] * v[q]);
            s[q] = (__bf16)(u[q] + v[q]);
        }
        *(v4bf*)&sE[wave][r][c]      = p;             // u*v  -> cols [0,128)
        *(v4bf*)&sE[wave][r][DD + c] = s;             // u+v  -> cols [128,256)
    }

    // Fragment lane mapping (ISA 16-bit A 16x32 and B 32x16 layouts)
    const int arow = lane & 15;
    const int asel = (lane >> 4) << 3;    // 0 or 8
    const int bn   = lane & 15;
    const int bko  = (lane >> 4) << 3;    // 0 or 8

    // ---- GEMM1: (16x256) @ (256x64), bf16 WMMA, f32 accum ----
    v8f acc0 = {}, acc1 = {}, acc2 = {}, acc3 = {};
    #pragma unroll
    for (int kk = 0; kk < 8; ++kk) {
        const int kb = kk * 32;
        AF a;
        a.h[0] = *(const v8bf*)&sE[wave][arow][kb + asel];
        a.h[1] = *(const v8bf*)&sE[wave][arow][kb + 16 + asel];
        AF w0, w1, w2, w3;
        const int bk = kb + bko;
        w0.h[0] = *(const v8bf*)&sW1t[bn     ][bk]; w0.h[1] = *(const v8bf*)&sW1t[bn     ][bk + 16];
        w1.h[0] = *(const v8bf*)&sW1t[bn + 16][bk]; w1.h[1] = *(const v8bf*)&sW1t[bn + 16][bk + 16];
        w2.h[0] = *(const v8bf*)&sW1t[bn + 32][bk]; w2.h[1] = *(const v8bf*)&sW1t[bn + 32][bk + 16];
        w3.h[0] = *(const v8bf*)&sW1t[bn + 48][bk]; w3.h[1] = *(const v8bf*)&sW1t[bn + 48][bk + 16];
        acc0 = __builtin_amdgcn_wmma_f32_16x16x32_bf16(false, a.v, false, w0.v, (short)0, acc0, false, false);
        acc1 = __builtin_amdgcn_wmma_f32_16x16x32_bf16(false, a.v, false, w1.v, (short)0, acc1, false, false);
        acc2 = __builtin_amdgcn_wmma_f32_16x16x32_bf16(false, a.v, false, w2.v, (short)0, acc2, false, false);
        acc3 = __builtin_amdgcn_wmma_f32_16x16x32_bf16(false, a.v, false, w3.v, (short)0, acc3, false, false);
    }
    // scatter D (C/D layout: vgpr r -> M=r (lanes 0-15) / M=8+r (lanes 16-31))
    {
        const int m  = (lane >> 4) << 3;
        const int nc = lane & 15;
        #pragma unroll
        for (int r = 0; r < 8; ++r) {
            sH[wave][m + r][nc     ] = acc0[r] + sB1[nc     ];
            sH[wave][m + r][nc + 16] = acc1[r] + sB1[nc + 16];
            sH[wave][m + r][nc + 32] = acc2[r] + sB1[nc + 32];
            sH[wave][m + r][nc + 48] = acc3[r] + sB1[nc + 48];
        }
    }

    // ---- LayerNorm(64) + exact GELU -> bf16 tile sA ----
    {
        const int row = lane & 15;
        float mu = 0.f;
        #pragma unroll
        for (int k = 0; k < HIDN; k += 4) {
            const v4f x = *(const v4f*)&sH[wave][row][k];
            mu += x[0] + x[1] + x[2] + x[3];
        }
        mu *= (1.0f / HIDN);
        float var = 0.f;
        #pragma unroll
        for (int k = 0; k < HIDN; k += 4) {
            const v4f x = *(const v4f*)&sH[wave][row][k];
            #pragma unroll
            for (int q = 0; q < 4; ++q) { const float d = x[q] - mu; var += d * d; }
        }
        const float rstd = rsqrtf(var * (1.0f / HIDN) + 1e-5f);
        const int c0 = (lane >> 4) * 32;            // lanes 0-15: cols 0-31, 16-31: 32-63
        #pragma unroll
        for (int k = 0; k < 32; k += 4) {
            v4bf o;
            #pragma unroll
            for (int q = 0; q < 4; ++q) {
                const int cc = c0 + k + q;
                const float xv = (sH[wave][row][cc] - mu) * rstd * sG1[cc] + sBe1[cc];
                o[q] = (__bf16)gelu_exact(xv);
            }
            *(v4bf*)&sA[wave][row][c0 + k] = o;
        }
    }

    // ---- GEMM2: (16x64) @ (64x32) ----
    v8f d0 = {}, d1 = {};
    #pragma unroll
    for (int kk = 0; kk < 2; ++kk) {
        const int kb = kk * 32;
        AF a;
        a.h[0] = *(const v8bf*)&sA[wave][arow][kb + asel];
        a.h[1] = *(const v8bf*)&sA[wave][arow][kb + 16 + asel];
        AF w0, w1;
        const int bk = kb + bko;
        w0.h[0] = *(const v8bf*)&sW2t[bn     ][bk]; w0.h[1] = *(const v8bf*)&sW2t[bn     ][bk + 16];
        w1.h[0] = *(const v8bf*)&sW2t[bn + 16][bk]; w1.h[1] = *(const v8bf*)&sW2t[bn + 16][bk + 16];
        d0 = __builtin_amdgcn_wmma_f32_16x16x32_bf16(false, a.v, false, w0.v, (short)0, d0, false, false);
        d1 = __builtin_amdgcn_wmma_f32_16x16x32_bf16(false, a.v, false, w1.v, (short)0, d1, false, false);
    }
    {
        const int m  = (lane >> 4) << 3;
        const int nc = lane & 15;
        #pragma unroll
        for (int r = 0; r < 8; ++r) {
            sH[wave][m + r][nc     ] = d0[r] + sB2[nc     ];
            sH[wave][m + r][nc + 16] = d1[r] + sB2[nc + 16];
        }
    }

    // ---- LayerNorm(32) + GELU -> f32 at cols [32,64) of sH ----
    {
        const int row = lane & 15;
        float mu = 0.f;
        #pragma unroll
        for (int k = 0; k < HID2; k += 4) {
            const v4f x = *(const v4f*)&sH[wave][row][k];
            mu += x[0] + x[1] + x[2] + x[3];
        }
        mu *= (1.0f / HID2);
        float var = 0.f;
        #pragma unroll
        for (int k = 0; k < HID2; k += 4) {
            const v4f x = *(const v4f*)&sH[wave][row][k];
            #pragma unroll
            for (int q = 0; q < 4; ++q) { const float d = x[q] - mu; var += d * d; }
        }
        const float rstd = rsqrtf(var * (1.0f / HID2) + 1e-5f);
        const int c0 = (lane >> 4) * 16;            // lanes 0-15: cols 0-15, 16-31: 16-31
        #pragma unroll
        for (int k = 0; k < 16; ++k) {
            const int cc = c0 + k;
            const float xv = (sH[wave][row][cc] - mu) * rstd * sG2[cc] + sBe2[cc];
            sH[wave][row][32 + cc] = gelu_exact(xv);
        }
    }

    // ---- GEMM3: (16x32) @ (32x1) -> output ----
    if (lane < 16) {
        const int row = lane;
        float acc = sB3;
        #pragma unroll
        for (int k = 0; k < HID2; k += 4) {
            const v4f x = *(const v4f*)&sH[wave][row][32 + k];
            const v4f w = *(const v4f*)&sW3[k];
            acc += x[0] * w[0] + x[1] * w[1] + x[2] * w[2] + x[3] * w[3];
        }
        const int e = ebase + row;
        if (e < E) out[(size_t)bidx * E + e] = acc;
    }
}

extern "C" void kernel_launch(void* const* d_in, const int* in_sizes, int n_in,
                              void* d_out, int out_size, void* d_ws, size_t ws_size,
                              hipStream_t stream) {
    const float* H   = (const float*)d_in[0];
    const int*   idx = (const int*)  d_in[1];   // JAX default x64-disabled -> int32
    const float* W1  = (const float*)d_in[2];
    const float* b1  = (const float*)d_in[3];
    const float* g1  = (const float*)d_in[4];
    const float* be1 = (const float*)d_in[5];
    const float* W2  = (const float*)d_in[6];
    const float* b2  = (const float*)d_in[7];
    const float* g2  = (const float*)d_in[8];
    const float* be2 = (const float*)d_in[9];
    const float* W3  = (const float*)d_in[10];
    const float* b3  = (const float*)d_in[11];

    const int E = in_sizes[1] / 2;                      // 35778
    const int B = in_sizes[0] / (NROW * DD);            // 32

    dim3 grid((E + EPB - 1) / EPB, B);
    edge_mlp_wmma_kernel<<<grid, dim3(256), 0, stream>>>(
        H, idx, W1, b1, g1, be1, W2, b2, g2, be2, W3, b3, (float*)d_out, E);
}
